// GeometricFeatureExtractor_88622355186348
// MI455X (gfx1250) — matrix operand order repaired and checked by reading, not compiled, not run on hardware
//
#include <hip/hip_runtime.h>
#include <hip/hip_bf16.h>

typedef __attribute__((ext_vector_type(16))) _Float16 v16h;
typedef __attribute__((ext_vector_type(8)))  _Float16 v8h;
typedef __attribute__((ext_vector_type(4)))  _Float16 h4;
typedef __attribute__((ext_vector_type(8)))  float    v8f;

#define N_PTS      100000
#define HID        1024
#define HALF_      512
#define ROWS       80            // rows per block; 100000 % 80 == 0 (no guards)
#define LDS_STRIDE 528           // 512 + 16 halfs pad; row stride 1056 B (32B aligned)

// ---------------------------------------------------------------------------
// One-shot: w2 [512,1024] f32 (row-major) -> w2T [1024,512] f16 in workspace.
// ---------------------------------------------------------------------------
__global__ __launch_bounds__(256) void w2_transpose_f16(const float* __restrict__ w2,
                                                        _Float16* __restrict__ w2t) {
  int idx = blockIdx.x * 256 + threadIdx.x;     // 0 .. 512*1024-1
  int k = idx >> 10;                            // 0..511
  int n = idx & (HID - 1);                      // 0..1023
  w2t[n * HALF_ + k] = (_Float16)w2[idx];
}

// B fragment (16-bit B 32x16): lane holds K chunks [base..base+7], [base+16..base+23] of column lane16.
__device__ __forceinline__ v16h load_bfrag(const _Float16* __restrict__ p) {
  v8h lo = *(const v8h*)(p);
  v8h hi = *(const v8h*)(p + 16);
  return __builtin_shufflevector(lo, hi, 0,1,2,3,4,5,6,7,8,9,10,11,12,13,14,15);
}

// A fragment (16-bit A 16x32): lane holds K chunks [base..base+7], [base+16..base+23].
__device__ __forceinline__ v16h load_afrag(const _Float16* p) {
  v8h lo = *(const v8h*)(p);
  v8h hi = *(const v8h*)(p + 16);
  return __builtin_shufflevector(lo, hi, 0,1,2,3,4,5,6,7,8,9,10,11,12,13,14,15);
}

#define WMMA_F16(A, B, C) \
  __builtin_amdgcn_wmma_f32_16x16x32_f16(false, (A), false, (B), (short)0, (C), false, false)

// ---------------------------------------------------------------------------
// Fused: geo features + layer1 (fp32 VALU -> f16 LDS) + layer2 (WMMA f16->f32)
// grid = (100000/80, 3 sets), block = 256 (8 waves).
// Wave w owns columns [w*128, w*128+128); 4 tile-pairs x 5 row-groups.
// kdim is a kernel argument (=512): runtime trip count prevents full unroll /
// cross-tp LICM of A-fragment ds_loads (round-3/4 scratch-spill regression).
// Zero-init accumulators let the first WMMA use the inline-0 C operand; the
// epilogue bias adds co-execute under WMMA (XDL ops count as TRANS).
// ---------------------------------------------------------------------------
__global__ __launch_bounds__(256, 4) void geo_mlp_kernel(
    const float* __restrict__ pos0, const float* __restrict__ pos1,
    const float* __restrict__ pos2,
    const float* __restrict__ w1,  const float* __restrict__ b1,
    const _Float16* __restrict__ w2t, const float* __restrict__ b2,
    float* __restrict__ out, int kdim) {
  __shared__ _Float16 hA[ROWS * LDS_STRIDE];    // 84480 B (gfx1250 allows 320 KB/WG)

  const int set = blockIdx.y;
  const float* __restrict__ pos = (set == 0) ? pos0 : (set == 1) ? pos1 : pos2;
  float* __restrict__ outSet = out + (size_t)set * N_PTS * HID;
  const int blockRow = blockIdx.x * ROWS;

  // ---------------- Phase 1: h = relu(feats @ w1 + b1), f16 into LDS -------
  {
    const int t  = threadIdx.x;
    const int q  = t & 3;                       // 4 threads per row, 128 cols each
    const int j0 = q * 128;
    for (int rr = (t >> 2); rr < ROWS; rr += 64) {   // rows 0..63, then 64..79
      const int row = blockRow + rr;                 // exact tiling: row < N_PTS
      const float x = pos[row * 3 + 0];
      const float y = pos[row * 3 + 1];
      const float z = pos[row * 3 + 2];
      const float ss = x * x + y * y + z * z;
      const float f0 = sqrtf(ss);
      const float f1 = x + y + z;
      const float f2 = ss;
      #pragma unroll 8
      for (int j = j0; j < j0 + 128; j += 4) {
        const float4 wa = *(const float4*)&w1[j];
        const float4 wb = *(const float4*)&w1[HALF_ + j];
        const float4 wc = *(const float4*)&w1[2 * HALF_ + j];
        const float4 bb = *(const float4*)&b1[j];
        float h0 = fmaxf(fmaf(f0, wa.x, fmaf(f1, wb.x, fmaf(f2, wc.x, bb.x))), 0.0f);
        float h1 = fmaxf(fmaf(f0, wa.y, fmaf(f1, wb.y, fmaf(f2, wc.y, bb.y))), 0.0f);
        float h2v = fmaxf(fmaf(f0, wa.z, fmaf(f1, wb.z, fmaf(f2, wc.z, bb.z))), 0.0f);
        float h3 = fmaxf(fmaf(f0, wa.w, fmaf(f1, wb.w, fmaf(f2, wc.w, bb.w))), 0.0f);
        h4 p;
        p[0] = (_Float16)h0; p[1] = (_Float16)h1;
        p[2] = (_Float16)h2v; p[3] = (_Float16)h3;
        *(h4*)&hA[rr * LDS_STRIDE + j] = p;     // ds_store_b64
      }
    }
  }
  __syncthreads();

  // ---------------- Phase 2: out = h @ w2 + b2 via v_wmma_f32_16x16x32_f16 -
  const int lane   = threadIdx.x & 31;
  const int wave   = threadIdx.x >> 5;          // 0..7 -> 128-col strip
  const int lane16 = lane & 15;
  const int laneHi = lane >> 4;
  const int colWave = wave * 128;

  // Two LDS bases keep all DS immediates < 64 KB: rg0-2 off aBase, rg3-4 off aBaseHi.
  const _Float16* aBase   = &hA[lane16 * LDS_STRIDE + laneHi * 8];
  const _Float16* aBaseHi = aBase + 3 * 16 * LDS_STRIDE;

  #pragma unroll 1
  for (int tp = 0; tp < 4; ++tp) {              // 4 pairs of 16-col tiles
    const int c0 = colWave + tp * 32;
    const _Float16* bp0 = w2t + (size_t)(c0 + lane16) * HALF_ + laneHi * 8;
    const _Float16* bp1 = bp0 + 16 * HALF_;

    v8f a00 = {}, a01 = {}, a10 = {}, a11 = {}, a20 = {}, a21 = {};
    v8f a30 = {}, a31 = {}, a40 = {}, a41 = {};

    v16h bc0 = load_bfrag(bp0);                 // double-buffered B
    v16h bc1 = load_bfrag(bp1);

    const int kT = kdim - 32;                   // runtime bound: no full unroll
    #pragma unroll 2
    for (int k0 = 0; k0 < kT; k0 += 32) {
      v16h af0 = load_afrag(aBase + k0);
      v16h af1 = load_afrag(aBase + 1 * 16 * LDS_STRIDE + k0);
      v16h af2 = load_afrag(aBase + 2 * 16 * LDS_STRIDE + k0);
      v16h af3 = load_afrag(aBaseHi + k0);
      v16h af4 = load_afrag(aBaseHi + 1 * 16 * LDS_STRIDE + k0);

      v16h bu0 = bc0, bu1 = bc1;
      bc0 = load_bfrag(bp0 + k0 + 32);          // prefetch next K (branch-free)
      bc1 = load_bfrag(bp1 + k0 + 32);

      a00 = WMMA_F16(af0, bu0, a00);  a01 = WMMA_F16(af0, bu1, a01);
      a10 = WMMA_F16(af1, bu0, a10);  a11 = WMMA_F16(af1, bu1, a11);
      a20 = WMMA_F16(af2, bu0, a20);  a21 = WMMA_F16(af2, bu1, a21);
      a30 = WMMA_F16(af3, bu0, a30);  a31 = WMMA_F16(af3, bu1, a31);
      a40 = WMMA_F16(af4, bu0, a40);  a41 = WMMA_F16(af4, bu1, a41);
    }
    {                                           // peeled tail: k0 = kdim-32
      v16h af0 = load_afrag(aBase + kT);
      v16h af1 = load_afrag(aBase + 1 * 16 * LDS_STRIDE + kT);
      v16h af2 = load_afrag(aBase + 2 * 16 * LDS_STRIDE + kT);
      v16h af3 = load_afrag(aBaseHi + kT);
      v16h af4 = load_afrag(aBaseHi + 1 * 16 * LDS_STRIDE + kT);
      a00 = WMMA_F16(af0, bc0, a00);  a01 = WMMA_F16(af0, bc1, a01);
      a10 = WMMA_F16(af1, bc0, a10);  a11 = WMMA_F16(af1, bc1, a11);
      a20 = WMMA_F16(af2, bc0, a20);  a21 = WMMA_F16(af2, bc1, a21);
      a30 = WMMA_F16(af3, bc0, a30);  a31 = WMMA_F16(af3, bc1, a31);
      a40 = WMMA_F16(af4, bc0, a40);  a41 = WMMA_F16(af4, bc1, a41);
    }

    // Epilogue: C/D layout — VGPR j of lane l -> (M = j + 8*laneHi, N = lane16).
    const float bias0 = b2[c0 + lane16];
    const float bias1 = b2[c0 + 16 + lane16];

    auto store_tile = [&](v8f acc, int col, int rg, float bias) {
      const int rbase = blockRow + rg * 16 + laneHi * 8;
      #pragma unroll
      for (int j = 0; j < 8; ++j)
        __builtin_nontemporal_store(acc[j] + bias,
                                    &outSet[(size_t)(rbase + j) * HID + col]);
    };

    store_tile(a00, c0 + lane16,      0, bias0);
    store_tile(a01, c0 + 16 + lane16, 0, bias1);
    store_tile(a10, c0 + lane16,      1, bias0);
    store_tile(a11, c0 + 16 + lane16, 1, bias1);
    store_tile(a20, c0 + lane16,      2, bias0);
    store_tile(a21, c0 + 16 + lane16, 2, bias1);
    store_tile(a30, c0 + lane16,      3, bias0);
    store_tile(a31, c0 + 16 + lane16, 3, bias1);
    store_tile(a40, c0 + lane16,      4, bias0);
    store_tile(a41, c0 + 16 + lane16, 4, bias1);
  }
}

extern "C" void kernel_launch(void* const* d_in, const int* in_sizes, int n_in,
                              void* d_out, int out_size, void* d_ws, size_t ws_size,
                              hipStream_t stream) {
  (void)in_sizes; (void)n_in; (void)out_size; (void)ws_size;
  const float* pos0 = (const float*)d_in[0];
  const float* pos1 = (const float*)d_in[1];
  const float* pos2 = (const float*)d_in[2];
  const float* w1   = (const float*)d_in[3];
  const float* b1   = (const float*)d_in[4];
  const float* w2   = (const float*)d_in[5];
  const float* b2   = (const float*)d_in[6];
  _Float16* w2t = (_Float16*)d_ws;              // 1 MB scratch

  hipLaunchKernelGGL(w2_transpose_f16, dim3((HALF_ * HID) / 256), dim3(256), 0, stream,
                     w2, w2t);

  dim3 grid(N_PTS / ROWS, 3);
  hipLaunchKernelGGL(geo_mlp_kernel, grid, dim3(256), 0, stream,
                     pos0, pos1, pos2, w1, b1, w2t, b2, (float*)d_out, HALF_);
}
